// R_GCN_42056319762602
// MI455X (gfx1250) — matrix operand order repaired and checked by reading, not compiled, not run on hardware
//
#include <hip/hip_runtime.h>
#include <hip/hip_bf16.h>

typedef __attribute__((ext_vector_type(16))) _Float16 v16h;
typedef __attribute__((ext_vector_type(8)))  float    v8f;

static constexpr int ROW_TILES = 10;   // 16-row tiles swept per wave (31250 = 3125*10)

// ---------------------------------------------------------------------------
// WMMA GEMM: C[M,F] = A[M,K] @ W[K,F]   (f32 mem, f16 multiply, f32 acc)
// One wave per (block.x row-stripe, block.y 16-col tile).
// B fragments (K/32 of them) are register-resident for the whole stripe.
//
// Register layouts used here (wave32; A/B k order is contiguous, not the ISA table order):
//   A 16x32 f16 : lane l -> row m=l%16, half=l/16; elements 0..7 hold
//                 k = 16*half+0..7, elements 8..15 hold k = 16*half+8..15
//                 => four contiguous float4 spans per lane.
//   B 32x16 f16 : lane l -> col n=l%16, half=l/16; element i holds k=16*half+i
//   C 16x16 f32 : lane l -> col n=l%16; vgpr r -> row m = r + 8*half
// ---------------------------------------------------------------------------
template<int K, int F>
__global__ __launch_bounds__(32) void gcn_gemm_wmma(
    const float* __restrict__ A, const float* __restrict__ W,
    float* __restrict__ C, int M)
{
    constexpr int KF = K / 32;
    const int lane = threadIdx.x & 31;
    const int half = lane >> 4;     // 0/1
    const int sub  = lane & 15;     // row (A) / col (B,C) within tile
    const int colBase = blockIdx.y << 4;
    const int col  = colBase + sub;
    const bool colOK = (col < F);   // folds to true when F is a multiple of 16

    // ---- Preload all B fragments for this column tile (stay in VGPRs) ----
    v16h bfrag[KF];
    #pragma unroll
    for (int kf = 0; kf < KF; ++kf) {
        if (colOK) {
            const float* bp = W + (size_t)(32 * kf + 16 * half) * F + col;
            #pragma unroll
            for (int i = 0; i < 16; ++i)
                bfrag[kf][i] = (_Float16)bp[(size_t)i * F];
        } else {
            #pragma unroll
            for (int i = 0; i < 16; ++i) bfrag[kf][i] = (_Float16)0.0f;
        }
    }

    const int tile0 = blockIdx.x * ROW_TILES;
    #pragma unroll 1
    for (int rt = 0; rt < ROW_TILES; ++rt) {
        const int rowBase = (tile0 + rt) << 4;
        if (rowBase >= M) break;
        const int arow = min(rowBase + sub, M - 1);
        const float* ap = A + (size_t)arow * K;

        v8f acc = {};
        #pragma unroll
        for (int kf = 0; kf < KF; ++kf) {
            const int kb0 = 32 * kf + 16 * half;       // elements 0..7
            const int kb1 = kb0 + 8;                   // elements 8..15
            const float4 q0 = *(const float4*)(ap + kb0);
            const float4 q1 = *(const float4*)(ap + kb0 + 4);
            const float4 q2 = *(const float4*)(ap + kb1);
            const float4 q3 = *(const float4*)(ap + kb1 + 4);
            v16h a;
            a[0]  = (_Float16)q0.x; a[1]  = (_Float16)q0.y;
            a[2]  = (_Float16)q0.z; a[3]  = (_Float16)q0.w;
            a[4]  = (_Float16)q1.x; a[5]  = (_Float16)q1.y;
            a[6]  = (_Float16)q1.z; a[7]  = (_Float16)q1.w;
            a[8]  = (_Float16)q2.x; a[9]  = (_Float16)q2.y;
            a[10] = (_Float16)q2.z; a[11] = (_Float16)q2.w;
            a[12] = (_Float16)q3.x; a[13] = (_Float16)q3.y;
            a[14] = (_Float16)q3.z; a[15] = (_Float16)q3.w;
            // EXEC is full here as WMMA requires.
            acc = __builtin_amdgcn_wmma_f32_16x16x32_f16(
                false, a, false, bfrag[kf], (short)0, acc, false, false);
        }

        // Prefetch next row tile's A stripe (global_prefetch_b8 path).
        if (rowBase + 16 < M)
            __builtin_prefetch(ap + (size_t)16 * K, 0, 3);

        if (colOK) {
            // Lane's outputs: column `col`, rows rowBase + 8*half + (0..7).
            // One base pointer + compile-time immediate offsets r*F.
            float* cp = C + (size_t)(rowBase + 8 * half) * F + col;
            if (rowBase + 16 <= M) {          // full tile (always, when M%16==0)
                #pragma unroll
                for (int r = 0; r < 8; ++r)
                    cp[(size_t)(r * F)] = acc[r];
            } else {                           // generic tail
                #pragma unroll
                for (int r = 0; r < 8; ++r)
                    if (rowBase + 8 * half + r < M)
                        cp[(size_t)(r * F)] = acc[r];
            }
        }
    }
}

// ---------------------------------------------------------------------------
// agg[row,f] += vals[e] * support[col,f]   one thread per (edge, feature)
// F = 1<<LOGF is a power of two -> shift/mask decomposition, coalesced within
// an edge; fp32 global atomics (global_atomic_add_f32).
// ---------------------------------------------------------------------------
template<int LOGF>
__global__ void gcn_scatter_add(
    const float* __restrict__ support, const float* __restrict__ vals,
    const int* __restrict__ rows, const int* __restrict__ cols,
    float* __restrict__ agg, int E)
{
    constexpr int F = 1 << LOGF;
    const long long t = (long long)blockIdx.x * blockDim.x + threadIdx.x;
    if (t >= ((long long)E << LOGF)) return;
    const int e = (int)(t >> LOGF);
    const int f = (int)(t & (F - 1));
    const int r = rows[e];
    const int c = cols[e];
    atomicAdd(&agg[((size_t)r << LOGF) + f], vals[e] * support[((size_t)c << LOGF) + f]);
}

__global__ void gcn_zero(float* __restrict__ p, long long n)
{
    const long long i = (long long)blockIdx.x * blockDim.x + threadIdx.x;
    if (i < n) p[i] = 0.0f;
}

// h[i] = act(h[i] + bias[i & (F-1)]);  act = leaky_relu(0.1) if RELU
template<int LOGF, int RELU>
__global__ void gcn_bias_act(float* __restrict__ h, const float* __restrict__ bias,
                             long long n)
{
    constexpr int F = 1 << LOGF;
    const long long i = (long long)blockIdx.x * blockDim.x + threadIdx.x;
    if (i >= n) return;
    float v = h[i] + bias[(int)(i & (F - 1))];
    if (RELU) v = (v >= 0.0f) ? v : 0.1f * v;
    h[i] = v;
}

// ---------------------------------------------------------------------------
template<int K, int LOGF, int RELU>
static void run_layer(const float* h, const float* W, const float* b,
                      float* support, float* aggOut,
                      const float* evals, const int* erow, const int* ecol,
                      int N, int E, hipStream_t stream)
{
    constexpr int F = 1 << LOGF;
    dim3 gg((N / 16 + ROW_TILES - 1) / ROW_TILES, (F + 15) / 16);
    gcn_gemm_wmma<K, F><<<gg, 32, 0, stream>>>(h, W, support, N);

    const long long nAgg = (long long)N << LOGF;
    gcn_zero<<<(unsigned)((nAgg + 255) / 256), 256, 0, stream>>>(aggOut, nAgg);

    const long long nSc = (long long)E << LOGF;
    gcn_scatter_add<LOGF><<<(unsigned)((nSc + 255) / 256), 256, 0, stream>>>(
        support, evals, erow, ecol, aggOut, E);

    gcn_bias_act<LOGF, RELU><<<(unsigned)((nAgg + 255) / 256), 256, 0, stream>>>(
        aggOut, b, nAgg);
}

extern "C" void kernel_launch(void* const* d_in, const int* in_sizes, int n_in,
                              void* d_out, int out_size, void* d_ws, size_t ws_size,
                              hipStream_t stream)
{
    const float* x    = (const float*)d_in[0];
    const float* evals= (const float*)d_in[1];
    const float* W1a  = (const float*)d_in[2];  const float* b1a = (const float*)d_in[3];
    const float* W1b  = (const float*)d_in[4];  const float* b1b = (const float*)d_in[5];
    const float* W2a  = (const float*)d_in[6];  const float* b2a = (const float*)d_in[7];
    const float* W2b  = (const float*)d_in[8];  const float* b2b = (const float*)d_in[9];
    const float* W3   = (const float*)d_in[10]; const float* b3  = (const float*)d_in[11];
    const int*   erow = (const int*)d_in[12];
    const int*   ecol = (const int*)d_in[13];
    float* out = (float*)d_out;

    const int N = in_sizes[0] / 32;
    const int E = in_sizes[1];

    float* buf0 = (float*)d_ws;                  // current features / agg
    float* buf1 = buf0 + (size_t)N * 128;        // support = h @ W

    // gc1: 32 -> 64 -> 128 (leaky_relu)
    run_layer<32, 6, 1>(x,    W1a, b1a, buf1, buf0, evals, erow, ecol, N, E, stream);
    run_layer<64, 7, 1>(buf0, W1b, b1b, buf1, buf0, evals, erow, ecol, N, E, stream);
    // gc2: 128 -> 64 -> 32 (leaky_relu)
    run_layer<128, 6, 1>(buf0, W2a, b2a, buf1, buf0, evals, erow, ecol, N, E, stream);
    run_layer<64, 5, 1>(buf0, W2b, b2b, buf1, buf0, evals, erow, ecol, N, E, stream);
    // gc3: 32 -> 2 (no activation), scatter straight into d_out
    run_layer<32, 1, 0>(buf0, W3,  b3,  buf1, out,  evals, erow, ecol, N, E, stream);
}
